// LlamaAttention_8272107012099
// MI455X (gfx1250) — matrix operand order repaired and checked by reading, not compiled, not run on hardware
//
#include <hip/hip_runtime.h>
#include <hip/hip_bf16.h>

// ---------------------------------------------------------------------------
// Llama attention layer for gfx1250 (MI455X), bf16 WMMA pipeline.
//   B=1, S=2048, D=2048, H=32, HK=8 (GQA groups=4), HD=64, scale = 1/8.
// Data movement: GLOBAL_LOAD_ASYNC_TO_LDS_B128 (ASYNCcnt) for tile staging.
// ---------------------------------------------------------------------------

#define S_   2048
#define D_   2048
#define H_   32
#define HK_  8
#define HD_  64
#define GRP_ 4

typedef __attribute__((ext_vector_type(16))) __bf16 bf16x16;
typedef __attribute__((ext_vector_type(8)))  __bf16 bf16x8;
typedef __attribute__((ext_vector_type(8)))  float  f32x8;

// ---------------------------------------------------------------------------
// CDNA5 async copy: global -> LDS, 16 bytes per lane, tracked by ASYNCcnt.
// ---------------------------------------------------------------------------
__device__ __forceinline__ void async_copy_b128(void* lds_dst, const void* gsrc) {
    auto ldst = (__attribute__((address_space(3))) void*)lds_dst;
    asm volatile("global_load_async_to_lds_b128 %0, %1, off"
                 :
                 : "v"(ldst), "v"(gsrc)
                 : "memory");
}

__device__ __forceinline__ void wait_async0() {
    asm volatile("s_wait_asynccnt 0x0" ::: "memory");
}

// ---------------------------------------------------------------------------
// fp32 -> bf16 elementwise convert
// ---------------------------------------------------------------------------
__global__ void cvt_bf16(const float* __restrict__ src, __bf16* __restrict__ dst, int n) {
    int i = blockIdx.x * blockDim.x + threadIdx.x;
    if (i < n) dst[i] = (__bf16)src[i];
}

// ---------------------------------------------------------------------------
// C[M,N] (f32) = A[M,K] (bf16, row-major) * B[N,K]^T (bf16, row-major weights)
// Block: 128 threads (4 waves), 128x64 C tile, K chunked by 32 through LDS.
// Each wave owns 32 rows (two 16-row WMMA sub-tiles) x 64 cols -> 8 WMMAs/chunk.
// WMMA v_wmma_f32_16x16x32_bf16 fragment layouts per CDNA5 ISA 7.12.2:
//   A: lane L holds row M=L%16; halves 0..7 -> K=(L/16)*8..+7, halves 8..15 -> +16
//   B: lane L holds col N=L%16; halves 0..7 -> K=(L/16)*8..+7, halves 8..15 -> +16
//   C: lane L, vgpr r -> (M = r + 8*(L/16), N = L%16)
// ---------------------------------------------------------------------------
#define BM 128
#define BN 64
#define BK 32

__global__ __launch_bounds__(128) void gemm_bf16_nt(const __bf16* __restrict__ A,
                                                    const __bf16* __restrict__ B,
                                                    float* __restrict__ C,
                                                    int M, int N, int K) {
    __shared__ __align__(16) __bf16 As[BM * BK];   // 8KB
    __shared__ __align__(16) __bf16 Bs[BN * BK];   // 4KB

    const int tid  = threadIdx.x;
    const int wave = tid >> 5;
    const int lane = tid & 31;
    const int lo   = lane & 15;
    const int kg   = lane >> 4;

    const int mBase = blockIdx.y * BM;
    const int nBase = blockIdx.x * BN;

    f32x8 acc[2][4] = {};

    const int bRow = tid >> 1;
    const int bCol = (tid & 1) * 16;

    for (int k0 = 0; k0 < K; k0 += BK) {
        // stage A (128x32) and B (64x32) tiles via async global->LDS copies
        {
            const __bf16* ga = A + (size_t)(mBase + tid) * K + k0;
#pragma unroll
            for (int i = 0; i < 4; ++i)
                async_copy_b128(&As[tid * BK + i * 8], ga + i * 8);
            const __bf16* gb = B + (size_t)(nBase + bRow) * K + k0 + bCol;
            async_copy_b128(&Bs[bRow * BK + bCol],     gb);
            async_copy_b128(&Bs[bRow * BK + bCol + 8], gb + 8);
        }
        wait_async0();
        __syncthreads();

#pragma unroll
        for (int ms = 0; ms < 2; ++ms) {
            bf16x16 a;
            {
                const int m_l = wave * 32 + ms * 16 + lo;
                bf16x8 a0 = *(const bf16x8*)(&As[m_l * BK + kg * 8]);
                bf16x8 a1 = *(const bf16x8*)(&As[m_l * BK + 16 + kg * 8]);
#pragma unroll
                for (int i = 0; i < 8; ++i) { a[i] = a0[i]; a[8 + i] = a1[i]; }
            }
#pragma unroll
            for (int nt = 0; nt < 4; ++nt) {
                const int n_l = nt * 16 + lo;
                bf16x8 b0 = *(const bf16x8*)(&Bs[n_l * BK + kg * 8]);
                bf16x8 b1 = *(const bf16x8*)(&Bs[n_l * BK + 16 + kg * 8]);
                bf16x16 b;
#pragma unroll
                for (int i = 0; i < 8; ++i) { b[i] = b0[i]; b[8 + i] = b1[i]; }
                acc[ms][nt] = __builtin_amdgcn_wmma_f32_16x16x32_bf16(
                    false, a, false, b, (short)0, acc[ms][nt], false, false);
            }
        }
        __syncthreads();
    }

#pragma unroll
    for (int ms = 0; ms < 2; ++ms) {
#pragma unroll
        for (int nt = 0; nt < 4; ++nt) {
#pragma unroll
            for (int r = 0; r < 8; ++r) {
                const int row = mBase + wave * 32 + ms * 16 + r + 8 * kg;
                const int col = nBase + nt * 16 + lo;
                C[(size_t)row * N + col] = acc[ms][nt][r];
            }
        }
    }
}

// ---------------------------------------------------------------------------
// RoPE on f32 GEMM output -> bf16, head-major layout [h][s][64]
// ---------------------------------------------------------------------------
__global__ void rope_bf16(const float* __restrict__ src, __bf16* __restrict__ dst,
                          const float* __restrict__ cs, const float* __restrict__ sn,
                          int nh) {
    int idx = blockIdx.x * blockDim.x + threadIdx.x;
    int total = S_ * nh * 32;
    if (idx >= total) return;
    int s   = idx / (nh * 32);
    int rem = idx - s * (nh * 32);
    int h   = rem >> 5;
    int d   = rem & 31;
    float xr = src[(size_t)s * nh * 64 + h * 64 + d];
    float xi = src[(size_t)s * nh * 64 + h * 64 + d + 32];
    float c  = cs[s * 32 + d];
    float si = sn[s * 32 + d];
    dst[(size_t)h * S_ * 64 + s * 64 + d]      = (__bf16)(xr * c - xi * si);
    dst[(size_t)h * S_ * 64 + s * 64 + d + 32] = (__bf16)(xr * si + xi * c);
}

// V: f32 [s][HK*64] -> bf16 [h][s][64]
__global__ void vconv_bf16(const float* __restrict__ src, __bf16* __restrict__ dst) {
    int idx = blockIdx.x * blockDim.x + threadIdx.x;
    if (idx >= S_ * HK_ * 64) return;
    int s   = idx / (HK_ * 64);
    int rem = idx - s * (HK_ * 64);
    int h   = rem >> 6;
    int d   = rem & 63;
    dst[(size_t)h * S_ * 64 + s * 64 + d] = (__bf16)src[idx];
}

// ---------------------------------------------------------------------------
// Flash attention: one block per (head, 64-query tile). 4 waves, each owns
// 16 query rows. 32-key chunks; online softmax; causal mask computed inline.
//   Q,K,V bf16 head-major [h][s][64]; Y bf16 [s][H*64].
// ---------------------------------------------------------------------------
__global__ __launch_bounds__(128) void attn_flash(const __bf16* __restrict__ Qr,
                                                  const __bf16* __restrict__ Kr,
                                                  const __bf16* __restrict__ Vh,
                                                  __bf16* __restrict__ Y) {
    const int h  = blockIdx.y;
    const int qb = blockIdx.x * 64;
    const int hk = h >> 2;   // GQA: 4 query heads share one kv head

    __shared__ __align__(16) __bf16 Qs[64 * 64];     // query tile, row-major
    __shared__ __align__(16) __bf16 Ks[32 * 64];     // key tile, row-major [key][d]
    __shared__ __align__(16) __bf16 Vt[64 * 32];     // value tile transposed [d][key]
    __shared__ __align__(16) __bf16 Ps[4][16 * 32];  // per-wave P tile [m][key]

    const int tid  = threadIdx.x;
    const int wave = tid >> 5;
    const int lane = tid & 31;
    const int lo   = lane & 15;
    const int kg   = lane >> 4;

    {   // stage Q tile (64x64) via async copies
        const int row = tid >> 1;
        const int cb  = (tid & 1) * 32;
        const __bf16* g = Qr + (size_t)h * S_ * 64 + (size_t)(qb + row) * 64 + cb;
#pragma unroll
        for (int i = 0; i < 4; ++i)
            async_copy_b128(&Qs[row * 64 + cb + i * 8], g + i * 8);
    }
    wait_async0();
    __syncthreads();

    // A-fragments for Q (k-dim 64 split into two 32-chunks), kept in VGPRs
    bf16x16 aq[2];
    {
        const int m_l = wave * 16 + lo;
#pragma unroll
        for (int c = 0; c < 2; ++c) {
            bf16x8 x0 = *(const bf16x8*)(&Qs[m_l * 64 + c * 32 + kg * 8]);
            bf16x8 x1 = *(const bf16x8*)(&Qs[m_l * 64 + c * 32 + 16 + kg * 8]);
#pragma unroll
            for (int i = 0; i < 8; ++i) { aq[c][i] = x0[i]; aq[c][8 + i] = x1[i]; }
        }
    }

    float m_run[8], l_run[8];
    f32x8 o[4] = {};
#pragma unroll
    for (int r = 0; r < 8; ++r) { m_run[r] = -1e30f; l_run[r] = 0.0f; }

    const int nchunks = (qb >> 5) + 2;   // keys [0, qb+64) covered causally

    for (int jc = 0; jc < nchunks; ++jc) {
        const int j0 = jc * 32;
        {   // stage K (async, row-major) and V (manual, transposed) tiles
            const int kr = tid >> 2;
            const int cb = (tid & 3) * 16;
            const __bf16* gk = Kr + (size_t)hk * S_ * 64 + (size_t)(j0 + kr) * 64 + cb;
            async_copy_b128(&Ks[kr * 64 + cb],     gk);
            async_copy_b128(&Ks[kr * 64 + cb + 8], gk + 8);
            const __bf16* gv = Vh + (size_t)hk * S_ * 64 + (size_t)(j0 + kr) * 64 + cb;
            bf16x8 v0 = *(const bf16x8*)(gv);
            bf16x8 v1 = *(const bf16x8*)(gv + 8);
#pragma unroll
            for (int i = 0; i < 8; ++i) {
                Vt[(cb + i) * 32 + kr]     = v0[i];
                Vt[(cb + 8 + i) * 32 + kr] = v1[i];
            }
        }
        wait_async0();
        __syncthreads();

        // S = Q K^T : two 16-key column tiles, k-dim 64 = 2 WMMAs each
        f32x8 sacc[2];
#pragma unroll
        for (int kt = 0; kt < 2; ++kt) {
            f32x8 z = {};
#pragma unroll
            for (int c = 0; c < 2; ++c) {
                const int n_l = kt * 16 + lo;
                bf16x8 b0 = *(const bf16x8*)(&Ks[n_l * 64 + c * 32 + kg * 8]);
                bf16x8 b1 = *(const bf16x8*)(&Ks[n_l * 64 + c * 32 + 16 + kg * 8]);
                bf16x16 b;
#pragma unroll
                for (int i = 0; i < 8; ++i) { b[i] = b0[i]; b[8 + i] = b1[i]; }
                z = __builtin_amdgcn_wmma_f32_16x16x32_bf16(
                    false, aq[c], false, b, (short)0, z, false, false);
            }
            sacc[kt] = z;
        }

        // online softmax (row = r + 8*kg, key = j0 + kt*16 + lo)
        float p0[8], p1[8];
#pragma unroll
        for (int r = 0; r < 8; ++r) {
            const int qrow = qb + wave * 16 + r + 8 * kg;
            float s0 = sacc[0][r] * 0.125f;
            float s1 = sacc[1][r] * 0.125f;
            if (j0 + lo      > qrow) s0 = -1e30f;
            if (j0 + 16 + lo > qrow) s1 = -1e30f;
            float mx = fmaxf(s0, s1);
#pragma unroll
            for (int off = 1; off < 16; off <<= 1)
                mx = fmaxf(mx, __shfl_xor(mx, off, 32));
            const float mn = fmaxf(m_run[r], mx);
            const float f  = __expf(m_run[r] - mn);
            const float e0 = __expf(s0 - mn);
            const float e1 = __expf(s1 - mn);
            float sum = e0 + e1;
#pragma unroll
            for (int off = 1; off < 16; off <<= 1)
                sum += __shfl_xor(sum, off, 32);
            l_run[r] = l_run[r] * f + sum;
            m_run[r] = mn;
            p0[r] = e0;
            p1[r] = e1;
#pragma unroll
            for (int nt = 0; nt < 4; ++nt) o[nt][r] *= f;
        }

        // C-layout -> A-layout for P via per-wave LDS tile
#pragma unroll
        for (int r = 0; r < 8; ++r) {
            Ps[wave][(r + 8 * kg) * 32 + lo]      = (__bf16)p0[r];
            Ps[wave][(r + 8 * kg) * 32 + 16 + lo] = (__bf16)p1[r];
        }
        __syncthreads();

        bf16x16 pa;
        {
            bf16x8 x0 = *(const bf16x8*)(&Ps[wave][lo * 32 + kg * 8]);
            bf16x8 x1 = *(const bf16x8*)(&Ps[wave][lo * 32 + 16 + kg * 8]);
#pragma unroll
            for (int i = 0; i < 8; ++i) { pa[i] = x0[i]; pa[8 + i] = x1[i]; }
        }

        // O += P V : 4 d-tiles of 16, k-dim = 32 keys
#pragma unroll
        for (int nt = 0; nt < 4; ++nt) {
            bf16x8 b0 = *(const bf16x8*)(&Vt[(nt * 16 + lo) * 32 + kg * 8]);
            bf16x8 b1 = *(const bf16x8*)(&Vt[(nt * 16 + lo) * 32 + 16 + kg * 8]);
            bf16x16 b;
#pragma unroll
            for (int i = 0; i < 8; ++i) { b[i] = b0[i]; b[8 + i] = b1[i]; }
            o[nt] = __builtin_amdgcn_wmma_f32_16x16x32_bf16(
                false, pa, false, b, (short)0, o[nt], false, false);
        }
        __syncthreads();
    }

    // normalize and write Y[s][h*64+d] (bf16, feeds the output GEMM)
#pragma unroll
    for (int nt = 0; nt < 4; ++nt) {
#pragma unroll
        for (int r = 0; r < 8; ++r) {
            const int row = qb + wave * 16 + r + 8 * kg;
            const int col = h * 64 + nt * 16 + lo;
            Y[(size_t)row * (H_ * HD_) + col] = (__bf16)(o[nt][r] / l_run[r]);
        }
    }
}

// ---------------------------------------------------------------------------
// Launch
// ---------------------------------------------------------------------------
extern "C" void kernel_launch(void* const* d_in, const int* in_sizes, int n_in,
                              void* d_out, int out_size, void* d_ws, size_t ws_size,
                              hipStream_t stream) {
    (void)in_sizes; (void)n_in; (void)out_size; (void)ws_size;

    const float* hs   = (const float*)d_in[0];
    const float* fcos = (const float*)d_in[1];
    const float* fsin = (const float*)d_in[2];
    // d_in[3] atten_mask: causal mask is recomputed inline in attn_flash
    const float* Wq   = (const float*)d_in[4];
    const float* Wk   = (const float*)d_in[5];
    const float* Wv   = (const float*)d_in[6];
    const float* Wo   = (const float*)d_in[7];

    char* ws = (char*)d_ws;
    const size_t MB = 1ull << 20;
    __bf16* hs_bf = (__bf16*)(ws);            // 8MB; reused later as rotated Q
    __bf16* wq_bf = (__bf16*)(ws + 8  * MB);  // 8MB; reused later as Y
    __bf16* wk_bf = (__bf16*)(ws + 16 * MB);  // 2MB; reused later as rotated K
    __bf16* wv_bf = (__bf16*)(ws + 18 * MB);  // 2MB; reused later as V (head-major)
    __bf16* wo_bf = (__bf16*)(ws + 20 * MB);  // 8MB
    float*  q_f   = (float*)(ws + 28 * MB);   // 16MB
    float*  k_f   = (float*)(ws + 44 * MB);   // 4MB
    float*  v_f   = (float*)(ws + 48 * MB);   // 4MB   (peak 52MB)

    // 1) fp32 -> bf16
    cvt_bf16<<<(S_ * D_) / 256, 256, 0, stream>>>(hs, hs_bf, S_ * D_);
    cvt_bf16<<<(H_ * HD_ * D_) / 256, 256, 0, stream>>>(Wq, wq_bf, H_ * HD_ * D_);
    cvt_bf16<<<(HK_ * HD_ * D_) / 256, 256, 0, stream>>>(Wk, wk_bf, HK_ * HD_ * D_);
    cvt_bf16<<<(HK_ * HD_ * D_) / 256, 256, 0, stream>>>(Wv, wv_bf, HK_ * HD_ * D_);
    cvt_bf16<<<(D_ * H_ * HD_) / 256, 256, 0, stream>>>(Wo, wo_bf, D_ * H_ * HD_);

    // 2) QKV projections (WMMA GEMMs)
    gemm_bf16_nt<<<dim3((H_ * HD_) / BN, S_ / BM), 128, 0, stream>>>(
        hs_bf, wq_bf, q_f, S_, H_ * HD_, D_);
    gemm_bf16_nt<<<dim3((HK_ * HD_) / BN, S_ / BM), 128, 0, stream>>>(
        hs_bf, wk_bf, k_f, S_, HK_ * HD_, D_);
    gemm_bf16_nt<<<dim3((HK_ * HD_) / BN, S_ / BM), 128, 0, stream>>>(
        hs_bf, wv_bf, v_f, S_, HK_ * HD_, D_);

    // 3) RoPE + layout transform to head-major bf16 (reuses freed buffers)
    __bf16* qr = hs_bf;
    __bf16* kr = wk_bf;
    __bf16* vh = wv_bf;
    __bf16* yb = wq_bf;
    rope_bf16<<<(S_ * H_ * 32 + 255) / 256, 256, 0, stream>>>(q_f, qr, fcos, fsin, H_);
    rope_bf16<<<(S_ * HK_ * 32 + 255) / 256, 256, 0, stream>>>(k_f, kr, fcos, fsin, HK_);
    vconv_bf16<<<(S_ * HK_ * 64 + 255) / 256, 256, 0, stream>>>(v_f, vh);

    // 4) causal flash attention (WMMA QK^T and PV, online softmax)
    attn_flash<<<dim3(S_ / 64, H_), 128, 0, stream>>>(qr, kr, vh, yb);

    // 5) output projection -> d_out (f32)
    gemm_bf16_nt<<<dim3(D_ / BN, S_ / BM), 128, 0, stream>>>(
        yb, wo_bf, (float*)d_out, S_, D_, H_ * HD_);
}
